// SSSD_2259152798073
// MI455X (gfx1250) — matrix operand order repaired and checked by reading, not compiled, not run on hardware
//
#include <hip/hip_runtime.h>
#include <hip/hip_bf16.h>
#include <math.h>

// ---------------- problem constants ----------------
#define B_   8
#define KIN_ 32
#define L_   2048
#define C_   128
#define H_   256      // 2*C
#define NB_  6
#define NMODE 16
#define TCH  64       // scan chunk length
#define NCH  32       // L_/TCH chunks

typedef __attribute__((ext_vector_type(16))) __bf16       v16bf;
typedef __attribute__((ext_vector_type(8)))  float        v8f;
typedef __attribute__((ext_vector_type(8)))  unsigned int v8u;

__device__ __forceinline__ unsigned short f2bf(float x) {
  union { float f; unsigned u; } v; v.f = x;
  unsigned r = v.u + 0x7FFFu + ((v.u >> 16) & 1u);
  return (unsigned short)(r >> 16);
}

// =====================================================================
// Implicit-GEMM Conv1d (k=1 or 3, 'same' pad) + bias, via bf16 WMMA.
// Y(b,co,l) = sum_{ci,dk} W(co,ci,dk) * (X(b,ci,l+dk-pad) + chanAdd(b,ci))
// Tile: 64(Cout) x 64(L) per workgroup, 4 waves, each wave 2x2 WMMA tiles.
// Double-buffered LDS staging: tile k+1 is fetched into registers while
// WMMA consumes tile k; tile k+2 rows are prefetched (global_prefetch_b8).
// =====================================================================
template<int CIN, int COUT, int KW>
__global__ void __launch_bounds__(128)
conv_gemm(const float* __restrict__ X, const float* __restrict__ W,
          const float* __restrict__ bias, const float* __restrict__ chanAdd,
          float* __restrict__ Y) {
  constexpr int KD  = CIN * KW;
  constexpr int PAD = (KW - 1) / 2;
  constexpr int NK  = KD / 32;
  __shared__ unsigned short As[2][64][34];
  __shared__ unsigned short Bs[2][64][34];
  const int b    = blockIdx.z;
  const int l0   = blockIdx.x * 64;
  const int co0  = blockIdx.y * 64;
  const int tid  = threadIdx.x;
  const int wave = tid >> 5;
  const int lane = tid & 31;
  const int half = lane >> 4;
  const int l16  = lane & 15;
  const int wm   = (wave >> 1) * 32;
  const int wn   = (wave & 1)  * 32;

  float ra[16], rb[16];

  auto ldA = [&](int k0) {
    #pragma unroll
    for (int r = 0; r < 16; r++) {
      int idx = tid + 128 * r;               // 2048 elements, coalesced in k
      int m = idx >> 5, kk = idx & 31;
      int co = co0 + m;
      ra[r] = (co < COUT) ? W[co * KD + k0 + kk] : 0.f;
    }
  };
  auto ldB = [&](int k0) {
    #pragma unroll
    for (int r = 0; r < 16; r++) {
      int idx = tid + 128 * r;               // coalesced in l
      int kk = idx >> 6, n = idx & 63;
      int k  = k0 + kk;
      int ci = k / KW, dk = k % KW;
      int l  = l0 + n + dk - PAD;
      float v = 0.f;
      if (l >= 0 && l < L_) {
        v = X[((long)b * CIN + ci) * L_ + l];
        if (chanAdd) v += chanAdd[b * CIN + ci];
      }
      rb[r] = v;
    }
  };
  auto stTile = [&](int buf) {
    #pragma unroll
    for (int r = 0; r < 16; r++) {
      int idx = tid + 128 * r;
      As[buf][idx >> 5][idx & 31] = f2bf(ra[r]);
      Bs[buf][idx & 63][idx >> 6] = f2bf(rb[r]);
    }
  };

  v8f acc[2][2];
  for (int i = 0; i < 2; i++)
    for (int j = 0; j < 2; j++)
      for (int e = 0; e < 8; e++) acc[i][j][e] = 0.f;

  ldA(0); ldB(0); stTile(0);
  __syncthreads();

  for (int it = 0; it < NK; ++it) {
    const int cur = it & 1;
    if (it + 1 < NK) { ldA((it + 1) * 32); ldB((it + 1) * 32); }
    if (it + 2 < NK) {
      int k  = (it + 2) * 32 + lane;
      int ci = k / KW;
      __builtin_prefetch(&X[((long)b * CIN + ci) * L_ + l0], 0, 1);
    }
    #pragma unroll
    for (int rt = 0; rt < 2; rt++) {
      v8u au;
      int m = wm + rt * 16 + l16;
      #pragma unroll
      for (int v = 0; v < 8; v++) {
        int kb = v * 2 + half * 16;
        au[v] = *(const unsigned int*)&As[cur][m][kb];
      }
      v16bf af = __builtin_bit_cast(v16bf, au);
      #pragma unroll
      for (int ct = 0; ct < 2; ct++) {
        v8u bu;
        int n = wn + ct * 16 + l16;
        #pragma unroll
        for (int v = 0; v < 8; v++) {
          int kb = v * 2 + half * 16;
          bu[v] = *(const unsigned int*)&Bs[cur][n][kb];
        }
        v16bf bfv = __builtin_bit_cast(v16bf, bu);
        acc[rt][ct] = __builtin_amdgcn_wmma_f32_16x16x32_bf16(
            false, af, false, bfv, (short)0, acc[rt][ct], false, false);
      }
    }
    if (it + 1 < NK) stTile((it + 1) & 1);   // other buffer: no race with readers
    __syncthreads();
  }

  #pragma unroll
  for (int rt = 0; rt < 2; rt++)
    #pragma unroll
    for (int ct = 0; ct < 2; ct++)
      #pragma unroll
      for (int v = 0; v < 8; v++) {
        int m = co0 + wm + rt * 16 + v + 8 * half;
        int n = l0  + wn + ct * 16 + l16;
        if (m < COUT)
          Y[((long)b * COUT + m) * L_ + n] = acc[rt][ct][v] + bias[m];
      }
}

// =====================================================================
// S4 tables: per channel h, 64 threads (t = 0..63).
//  KT[h][t]      = 2 Re sum_n Cc_n w_n^t          (Toeplitz kernel row)
//  A2[h][t][j]   = j<16 ?  2 Re(Cc_n w_n^{t+1})   (carry-correction coeffs, bf16)
//                :        -2 Im(Cc_n w_n^{t+1})
//  P [h][j][tau] = Re/Im(w_n^{63-tau})            (chunk-projection coeffs, bf16)
//  WT[h][n]      = w_n^{64}                        (chunk-to-chunk state decay)
// =====================================================================
__global__ void __launch_bounds__(64)
s4_tables(const float* __restrict__ log_dt, const float* __restrict__ A_re,
          const float* __restrict__ A_im,  const float* __restrict__ C_re,
          const float* __restrict__ C_im,
          float* __restrict__ KT, unsigned short* __restrict__ A2,
          unsigned short* __restrict__ P, float* __restrict__ WT) {
  int h = blockIdx.x;
  int t = threadIdx.x;                 // 0..63
  float dt = expf(log_dt[h]);
  float ksum = 0.f;
  for (int n = 0; n < NMODE; n++) {
    float ar = A_re[h * NMODE + n], ai = A_im[h * NMODE + n];
    float adr = ar * dt, adi = ai * dt;
    // w = exp(dtA)
    float e1 = expf(adr), s1, c1; sincosf(adi, &s1, &c1);
    float w1r = e1 * c1, w1i = e1 * s1;
    // Cc = C * (w - 1)/A
    float inv = 1.f / (ar * ar + ai * ai);
    float nr = w1r - 1.f, ni = w1i;
    float qr = (nr * ar + ni * ai) * inv;
    float qi = (ni * ar - nr * ai) * inv;
    float cr = C_re[h * NMODE + n], ci = C_im[h * NMODE + n];
    float ccr = cr * qr - ci * qi;
    float cci = cr * qi + ci * qr;
    // w^t
    float et = expf(adr * t), st, ct; sincosf(adi * t, &st, &ct);
    float wtr = et * ct, wti = et * st;
    ksum += 2.f * (ccr * wtr - cci * wti);
    // w^{t+1}
    float e2 = expf(adr * (t + 1)), s2, c2; sincosf(adi * (t + 1), &s2, &c2);
    float w2r = e2 * c2, w2i = e2 * s2;
    A2[(h * TCH + t) * 32 + n]          = f2bf( 2.f * (ccr * w2r - cci * w2i));
    A2[(h * TCH + t) * 32 + 16 + n]     = f2bf(-2.f * (ccr * w2i + cci * w2r));
    // w^{63-t}
    float e3 = expf(adr * (63 - t)), s3, c3; sincosf(adi * (63 - t), &s3, &c3);
    P[(h * 32 + n)      * TCH + t] = f2bf(e3 * c3);
    P[(h * 32 + 16 + n) * TCH + t] = f2bf(e3 * s3);
    if (t == 0) {
      float e4 = expf(adr * 64.f), s4, c4; sincosf(adi * 64.f, &s4, &c4);
      WT[(h * NMODE + n) * 2]     = e4 * c4;
      WT[(h * NMODE + n) * 2 + 1] = e4 * s4;
    }
  }
  KT[h * TCH + t] = ksum;
}

// =====================================================================
// S4 chunk projection: a[b,h,c,j] = sum_tau P[h][j][tau] * u[b,h,c*64+tau]
// GEMM per h:  (32 x 64) @ (64 x 256 chunk-cols).  2 waves, 32x64 tile.
// =====================================================================
__global__ void __launch_bounds__(64)
s4_chunkproj(const unsigned short* __restrict__ P, const float* __restrict__ U,
             float* __restrict__ ACH) {
  __shared__ unsigned short As[32][66];
  __shared__ unsigned short Bs[64][66];
  const int h   = blockIdx.x;
  const int g0  = blockIdx.y * 64;       // global chunk column = b*32 + c
  const int tid = threadIdx.x;
  const int wave = tid >> 5, lane = tid & 31;
  const int half = lane >> 4, l16 = lane & 15;
  const int wn = wave * 32;

  #pragma unroll
  for (int r = 0; r < 32; r++) {         // A: 32*64 bf16 copy
    int idx = tid + 64 * r;
    int j = idx >> 6, tau = idx & 63;
    As[j][tau] = P[(h * 32 + j) * TCH + tau];
  }
  #pragma unroll
  for (int r = 0; r < 64; r++) {         // B: u chunks
    int idx = tid + 64 * r;
    int n = idx >> 6, tau = idx & 63;    // n in 0..63
    int g = g0 + n, b = g >> 5, c = g & 31;
    Bs[n][tau] = f2bf(U[((long)b * H_ + h) * L_ + c * TCH + tau]);
  }
  __syncthreads();

  v8f acc[2][2];
  for (int i = 0; i < 2; i++) for (int j = 0; j < 2; j++)
    for (int e = 0; e < 8; e++) acc[i][j][e] = 0.f;

  #pragma unroll
  for (int s = 0; s < 2; s++) {          // K = 64 -> two WMMA steps
    #pragma unroll
    for (int rt = 0; rt < 2; rt++) {
      v8u au;
      int m = rt * 16 + l16;
      #pragma unroll
      for (int v = 0; v < 8; v++) {
        int kb = s * 32 + v * 2 + half * 16;
        au[v] = *(const unsigned int*)&As[m][kb];
      }
      v16bf af = __builtin_bit_cast(v16bf, au);
      #pragma unroll
      for (int ct = 0; ct < 2; ct++) {
        v8u bu;
        int n = wn + ct * 16 + l16;
        #pragma unroll
        for (int v = 0; v < 8; v++) {
          int kb = s * 32 + v * 2 + half * 16;
          bu[v] = *(const unsigned int*)&Bs[n][kb];
        }
        v16bf bfv = __builtin_bit_cast(v16bf, bu);
        acc[rt][ct] = __builtin_amdgcn_wmma_f32_16x16x32_bf16(
            false, af, false, bfv, (short)0, acc[rt][ct], false, false);
      }
    }
  }
  #pragma unroll
  for (int rt = 0; rt < 2; rt++)
    #pragma unroll
    for (int ct = 0; ct < 2; ct++)
      #pragma unroll
      for (int v = 0; v < 8; v++) {
        int j = rt * 16 + v + 8 * half;
        int g = g0 + wn + ct * 16 + l16, b = g >> 5, c = g & 31;
        ACH[(((long)b * H_ + h) * NCH + c) * 32 + j] = acc[rt][ct][v];
      }
}

// Sequential chunk-state scan: S_in[c] = wT*S_in[c-1] + a[c-1], S_in[0]=0.
__global__ void s4_scan(const float* __restrict__ ACH, const float* __restrict__ WT,
                        float* __restrict__ S) {
  int idx = blockIdx.x * blockDim.x + threadIdx.x;       // B*H*16
  if (idx >= B_ * H_ * NMODE) return;
  int n = idx & 15, h = (idx >> 4) & (H_ - 1), b = idx / (NMODE * H_);
  float wr = WT[(h * NMODE + n) * 2], wi = WT[(h * NMODE + n) * 2 + 1];
  const float* a = ACH + (((long)b * H_ + h) * NCH) * 32;
  float*       s = S   + (((long)b * H_ + h) * NCH) * 32;
  float sr = 0.f, si = 0.f;
  for (int c = 0; c < NCH; c++) {
    s[c * 32 + n]      = sr;
    s[c * 32 + 16 + n] = si;
    float ar = a[c * 32 + n], ai = a[c * 32 + 16 + n];
    float tr = wr * sr - wi * si + ar;
    si = wr * si + wi * sr + ai;
    sr = tr;
  }
}

// =====================================================================
// S4 main: per h — triangular-Toeplitz GEMM (K=64) + carry GEMM (K=32),
// then  y = ssm + (D+1)*u   (D skip + residual). 4 waves, 64x64 tile.
// =====================================================================
__global__ void __launch_bounds__(128)
s4_toeplitz(const float* __restrict__ KT, const unsigned short* __restrict__ A2,
            const float* __restrict__ S, const float* __restrict__ U,
            const float* __restrict__ Dvec, float* __restrict__ Yout) {
  __shared__ unsigned short As[64][34];
  __shared__ unsigned short Bs[64][34];
  __shared__ float kt[64];
  const int h  = blockIdx.x;
  const int g0 = blockIdx.y * 64;
  const int tid = threadIdx.x;
  const int wave = tid >> 5, lane = tid & 31;
  const int half = lane >> 4, l16 = lane & 15;
  const int wm = (wave >> 1) * 32, wn = (wave & 1) * 32;

  if (tid < 64) kt[tid] = KT[h * TCH + tid];

  v8f acc[2][2];
  for (int i = 0; i < 2; i++) for (int j = 0; j < 2; j++)
    for (int e = 0; e < 8; e++) acc[i][j][e] = 0.f;

  // ---- phase 1: triangular Toeplitz, K = 64 ----
  for (int k0 = 0; k0 < 64; k0 += 32) {
    __syncthreads();
    #pragma unroll
    for (int r = 0; r < 16; r++) {
      int idx = tid + 128 * r;
      int t = idx >> 5, kk = idx & 31;
      int d = t - (k0 + kk);
      As[t][kk] = (d >= 0) ? f2bf(kt[d]) : (unsigned short)0;
    }
    #pragma unroll
    for (int r = 0; r < 16; r++) {
      int idx = tid + 128 * r;
      int kk = idx >> 6, n = idx & 63;
      int g = g0 + n, b = g >> 5, c = g & 31;
      Bs[n][kk] = f2bf(U[((long)b * H_ + h) * L_ + c * TCH + k0 + kk]);
    }
    __syncthreads();
    #pragma unroll
    for (int rt = 0; rt < 2; rt++) {
      v8u au;
      int m = wm + rt * 16 + l16;
      #pragma unroll
      for (int v = 0; v < 8; v++) {
        int kb = v * 2 + half * 16;
        au[v] = *(const unsigned int*)&As[m][kb];
      }
      v16bf af = __builtin_bit_cast(v16bf, au);
      #pragma unroll
      for (int ct = 0; ct < 2; ct++) {
        v8u bu;
        int n = wn + ct * 16 + l16;
        #pragma unroll
        for (int v = 0; v < 8; v++) {
          int kb = v * 2 + half * 16;
          bu[v] = *(const unsigned int*)&Bs[n][kb];
        }
        v16bf bfv = __builtin_bit_cast(v16bf, bu);
        acc[rt][ct] = __builtin_amdgcn_wmma_f32_16x16x32_bf16(
            false, af, false, bfv, (short)0, acc[rt][ct], false, false);
      }
    }
  }

  // ---- phase 2: carry correction, K = 32 (rank-2N) ----
  __syncthreads();
  #pragma unroll
  for (int r = 0; r < 16; r++) {
    int idx = tid + 128 * r;
    int t = idx >> 5, j = idx & 31;
    As[t][j] = A2[(h * TCH + t) * 32 + j];
  }
  #pragma unroll
  for (int r = 0; r < 16; r++) {
    int idx = tid + 128 * r;
    int j = idx & 31, n = idx >> 5;
    int g = g0 + n, b = g >> 5, c = g & 31;
    Bs[n][j] = f2bf(S[(((long)b * H_ + h) * NCH + c) * 32 + j]);
  }
  __syncthreads();
  #pragma unroll
  for (int rt = 0; rt < 2; rt++) {
    v8u au;
    int m = wm + rt * 16 + l16;
    #pragma unroll
    for (int v = 0; v < 8; v++) {
      int kb = v * 2 + half * 16;
      au[v] = *(const unsigned int*)&As[m][kb];
    }
    v16bf af = __builtin_bit_cast(v16bf, au);
    #pragma unroll
    for (int ct = 0; ct < 2; ct++) {
      v8u bu;
      int n = wn + ct * 16 + l16;
      #pragma unroll
      for (int v = 0; v < 8; v++) {
        int kb = v * 2 + half * 16;
        bu[v] = *(const unsigned int*)&Bs[n][kb];
      }
      v16bf bfv = __builtin_bit_cast(v16bf, bu);
      acc[rt][ct] = __builtin_amdgcn_wmma_f32_16x16x32_bf16(
          false, af, false, bfv, (short)0, acc[rt][ct], false, false);
    }
  }

  // ---- epilogue: + (D+1)*u ----
  float dpo = Dvec[h] + 1.0f;
  #pragma unroll
  for (int rt = 0; rt < 2; rt++)
    #pragma unroll
    for (int ct = 0; ct < 2; ct++)
      #pragma unroll
      for (int v = 0; v < 8; v++) {
        int t = wm + rt * 16 + v + 8 * half;
        int g = g0 + wn + ct * 16 + l16, b = g >> 5, c = g & 31;
        long o = ((long)b * H_ + h) * L_ + c * TCH + t;
        Yout[o] = acc[rt][ct][v] + dpo * U[o];
      }
}

// ---------------- BN / LN / elementwise ----------------
__global__ void bn_stats(const float* __restrict__ Y, int Cout, float* __restrict__ st) {
  __shared__ float sh[256], sh2[256];
  int c = blockIdx.x, tid = threadIdx.x;
  float s = 0.f, ss = 0.f;
  for (int i = tid; i < B_ * L_; i += 256) {
    int b = i >> 11, l = i & (L_ - 1);
    float v = Y[((long)b * Cout + c) * L_ + l];
    s += v; ss += v * v;
  }
  sh[tid] = s; sh2[tid] = ss; __syncthreads();
  for (int o = 128; o > 0; o >>= 1) {
    if (tid < o) { sh[tid] += sh[tid + o]; sh2[tid] += sh2[tid + o]; }
    __syncthreads();
  }
  if (tid == 0) {
    float m = sh[0] * (1.f / (B_ * L_));
    float var = sh2[0] * (1.f / (B_ * L_)) - m * m;
    st[c * 2] = m; st[c * 2 + 1] = rsqrtf(var + 1e-5f);
  }
}

__global__ void bn_apply(const float* __restrict__ src, float* __restrict__ dst,
                         const float* __restrict__ st, const float* __restrict__ g,
                         const float* __restrict__ be, int Cout, int relu) {
  long i = (long)blockIdx.x * 256 + threadIdx.x;
  if (i >= (long)B_ * Cout * L_) return;
  int c = (int)((i >> 11) % Cout);
  float v = (src[i] - st[c * 2]) * st[c * 2 + 1] * g[c] + be[c];
  if (relu) v = fmaxf(v, 0.f);
  dst[i] = v;
}

__global__ void ln_apply(const float* __restrict__ src, float* __restrict__ dst,
                         const float* __restrict__ g, const float* __restrict__ be) {
  int i = blockIdx.x * blockDim.x + threadIdx.x;    // over B*L
  if (i >= B_ * L_) return;
  int b = i >> 11, l = i & (L_ - 1);
  const float* p = src + (long)b * H_ * L_ + l;
  float s = 0.f, ss = 0.f;
  for (int c = 0; c < H_; c++) { float v = p[c * L_]; s += v; ss += v * v; }
  float m = s * (1.f / H_);
  float inv = rsqrtf(ss * (1.f / H_) - m * m + 1e-5f);
  float* q = dst + (long)b * H_ * L_ + l;
  for (int c = 0; c < H_; c++) q[c * L_] = (p[c * L_] - m) * inv * g[c] + be[c];
}

__global__ void k_concat(const float* __restrict__ a, const float* __restrict__ m,
                         float* __restrict__ out) {
  long i = (long)blockIdx.x * 256 + threadIdx.x;     // B*64*L
  if (i >= (long)B_ * 2 * KIN_ * L_) return;
  int l = (int)(i & (L_ - 1)); int c = (int)((i >> 11) % (2 * KIN_)); int b = (int)(i / ((long)2 * KIN_ * L_));
  float v = (c < KIN_) ? a[((long)b * KIN_ + c) * L_ + l]
                       : m[((long)b * KIN_ + (c - KIN_)) * L_ + l];
  out[i] = v;
}

__global__ void k_add(float* __restrict__ a, const float* __restrict__ b, long n) {
  long i = (long)blockIdx.x * 256 + threadIdx.x;
  if (i < n) a[i] += b[i];
}
__global__ void k_scale(float* __restrict__ a, float s, long n) {
  long i = (long)blockIdx.x * 256 + threadIdx.x;
  if (i < n) a[i] *= s;
}
__global__ void k_fill0(float* __restrict__ a, long n) {
  long i = (long)blockIdx.x * 256 + threadIdx.x;
  if (i < n) a[i] = 0.f;
}
__global__ void k_gate(const float* __restrict__ hh, float* __restrict__ out) {
  long i = (long)blockIdx.x * 256 + threadIdx.x;     // B*C*L
  if (i >= (long)B_ * C_ * L_) return;
  int l = (int)(i & (L_ - 1)); int c = (int)((i >> 11) % C_); int b = (int)(i / ((long)C_ * L_));
  float gte = hh[((long)b * H_ + c) * L_ + l];
  float flt = hh[((long)b * H_ + C_ + c) * L_ + l];
  out[i] = (1.f / (1.f + expf(-gte))) * tanhf(flt);
}

// ---------------- embedding MLP (tiny) ----------------
__global__ void embed1(const float* __restrict__ tbl, const int* __restrict__ step,
                       const float* __restrict__ W1, const float* __restrict__ b1,
                       float* __restrict__ E1) {
  int b = blockIdx.x, j = threadIdx.x;              // 512 threads
  const float* e0 = tbl + step[b] * 128;
  float s = b1[j];
  for (int i = 0; i < 128; i++) s += e0[i] * W1[i * 512 + j];
  E1[b * 512 + j] = s / (1.f + expf(-s));
}
__global__ void embed2(const float* __restrict__ E1, const float* __restrict__ W2,
                       const float* __restrict__ b2, float* __restrict__ E2) {
  int b = blockIdx.x, j = threadIdx.x;
  float s = b2[j];
  for (int i = 0; i < 512; i++) s += E1[b * 512 + i] * W2[i * 512 + j];
  E2[b * 512 + j] = s / (1.f + expf(-s));
}
__global__ void eproj(const float* __restrict__ E2, const float* __restrict__ ew,
                      const float* __restrict__ eb, float* __restrict__ out) {
  int b = blockIdx.x, c = threadIdx.x;              // 128 threads
  float s = eb[c];
  for (int i = 0; i < 512; i++) s += E2[b * 512 + i] * ew[i * 128 + c];
  out[b * 128 + c] = s;
}

// =====================================================================
// Host side
// =====================================================================
extern "C" void kernel_launch(void* const* d_in, const int* in_sizes, int n_in,
                              void* d_out, int out_size, void* d_ws, size_t ws_size,
                              hipStream_t stream) {
  // ---- param leaf indices (jax tree_flatten: sorted dict keys) ----
  // block i (34 leaves): cond{b,be,g,w}=0..3, emb_b=4, emb_w=5, mid{..}=6..9,
  //   res{..}=10..13 (dead code, unused), s4a{A_im,A_re,C_im,C_re,D,ln_b,ln_g,log_dt}=14..21,
  //   s4b=22..29, skip{..}=30..33.  After 6 blocks (base 204):
  //   emb_b1=204, emb_b2=205, emb_table=206, emb_w1=207, emb_w2=208,
  //   out1{b,be,g,w}=209..212, out2=213..216, xt_conv=217..220,
  //   x_t=221, x_co=222, mask=223, diff_step=224.
  auto F = [&](int i) { return (const float*)d_in[i]; };
  const float* x_t  = F(221);
  const float* x_co = F(222);
  const float* mask = F(223);
  const int*   step = (const int*)d_in[224];

  // ---- workspace bump allocator ----
  char* wsb = (char*)d_ws; size_t off = 0;
  auto alloc = [&](size_t bytes) -> void* {
    void* p = wsb + off; off = (off + bytes + 255) & ~(size_t)255; return p;
  };
  float* XC  = (float*)alloc((size_t)B_ * 64 * L_ * 4);
  float* E1  = (float*)alloc((size_t)B_ * 512 * 4);
  float* E2  = (float*)alloc((size_t)B_ * 512 * 4);
  float* EPJ = (float*)alloc((size_t)NB_ * B_ * 128 * 4);
  float* H0  = (float*)alloc((size_t)B_ * C_ * L_ * 4);
  float* T1  = (float*)alloc((size_t)B_ * H_ * L_ * 4);
  float* T2  = (float*)alloc((size_t)B_ * H_ * L_ * 4);
  float* GF  = (float*)alloc((size_t)B_ * C_ * L_ * 4);
  float* SK  = (float*)alloc((size_t)B_ * C_ * L_ * 4);
  float* ST  = (float*)alloc((size_t)H_ * 2 * 4);
  float* KT  = (float*)alloc((size_t)H_ * TCH * 4);
  float* WT  = (float*)alloc((size_t)H_ * NMODE * 2 * 4);
  float* ACH = (float*)alloc((size_t)B_ * H_ * NCH * 32 * 4);
  float* SS  = (float*)alloc((size_t)B_ * H_ * NCH * 32 * 4);
  unsigned short* A2T = (unsigned short*)alloc((size_t)H_ * TCH * 32 * 2);
  unsigned short* PT  = (unsigned short*)alloc((size_t)H_ * 32 * TCH * 2);
  (void)ws_size; (void)in_sizes; (void)n_in; (void)out_size;

  const long nH  = (long)B_ * H_ * L_;
  const long nC  = (long)B_ * C_ * L_;
  const dim3 g256(32, 4, B_), g128(32, 2, B_), g32(32, 1, B_);

  // ---- input prep ----
  hipLaunchKernelGGL(k_concat, dim3((B_ * 64 * L_ + 255) / 256), dim3(256), 0, stream,
                     x_co, mask, XC);
  hipLaunchKernelGGL(embed1, dim3(B_), dim3(512), 0, stream, F(206), step, F(207), F(204), E1);
  hipLaunchKernelGGL(embed2, dim3(B_), dim3(512), 0, stream, E1, F(208), F(205), E2);
  for (int nb = 0; nb < NB_; nb++)
    hipLaunchKernelGGL(eproj, dim3(B_), dim3(128), 0, stream,
                       E2, F(nb * 34 + 5), F(nb * 34 + 4), EPJ + (size_t)nb * B_ * 128);

  // ---- h0 = relu(BN(conv3(x_t))) ----
  hipLaunchKernelGGL((conv_gemm<32, 128, 3>), g128, dim3(128), 0, stream,
                     x_t, F(220), F(217), (const float*)nullptr, H0);
  hipLaunchKernelGGL(bn_stats, dim3(128), dim3(256), 0, stream, H0, 128, ST);
  hipLaunchKernelGGL(bn_apply, dim3((unsigned)((nC + 255) / 256)), dim3(256), 0, stream,
                     H0, H0, ST, F(219), F(218), 128, 1);

  hipLaunchKernelGGL(k_fill0, dim3((unsigned)((nC + 255) / 256)), dim3(256), 0, stream, SK, nC);

  for (int nb = 0; nb < NB_; nb++) {
    const int bb = nb * 34;
    // mid conv (fused +eproj channel bias) + BN
    hipLaunchKernelGGL((conv_gemm<128, 256, 3>), g256, dim3(128), 0, stream,
                       H0, F(bb + 9), F(bb + 6), EPJ + (size_t)nb * B_ * 128, T1);
    hipLaunchKernelGGL(bn_stats, dim3(256), dim3(256), 0, stream, T1, 256, ST);
    hipLaunchKernelGGL(bn_apply, dim3((unsigned)((nH + 255) / 256)), dim3(256), 0, stream,
                       T1, T1, ST, F(bb + 8), F(bb + 7), 256, 0);
    // s4a: tables -> chunkproj -> scan -> toeplitz -> LN (T1 -> T2 -> T1)
    hipLaunchKernelGGL(s4_tables, dim3(H_), dim3(64), 0, stream,
                       F(bb + 21), F(bb + 15), F(bb + 14), F(bb + 17), F(bb + 16),
                       KT, A2T, PT, WT);
    hipLaunchKernelGGL(s4_chunkproj, dim3(H_, 4), dim3(64), 0, stream, PT, T1, ACH);
    hipLaunchKernelGGL(s4_scan, dim3((B_ * H_ * NMODE + 255) / 256), dim3(256), 0, stream,
                       ACH, WT, SS);
    hipLaunchKernelGGL(s4_toeplitz, dim3(H_, 4), dim3(128), 0, stream,
                       KT, A2T, SS, T1, F(bb + 18), T2);
    hipLaunchKernelGGL(ln_apply, dim3(64), dim3(256), 0, stream, T2, T1, F(bb + 20), F(bb + 19));
    // cond conv + BN, add
    hipLaunchKernelGGL((conv_gemm<64, 256, 1>), g256, dim3(128), 0, stream,
                       XC, F(bb + 3), F(bb + 0), (const float*)nullptr, T2);
    hipLaunchKernelGGL(bn_stats, dim3(256), dim3(256), 0, stream, T2, 256, ST);
    hipLaunchKernelGGL(bn_apply, dim3((unsigned)((nH + 255) / 256)), dim3(256), 0, stream,
                       T2, T2, ST, F(bb + 2), F(bb + 1), 256, 0);
    hipLaunchKernelGGL(k_add, dim3((unsigned)((nH + 255) / 256)), dim3(256), 0, stream, T1, T2, nH);
    // s4b (T1 -> T2, LN in place)
    hipLaunchKernelGGL(s4_tables, dim3(H_), dim3(64), 0, stream,
                       F(bb + 29), F(bb + 23), F(bb + 22), F(bb + 25), F(bb + 24),
                       KT, A2T, PT, WT);
    hipLaunchKernelGGL(s4_chunkproj, dim3(H_, 4), dim3(64), 0, stream, PT, T1, ACH);
    hipLaunchKernelGGL(s4_scan, dim3((B_ * H_ * NMODE + 255) / 256), dim3(256), 0, stream,
                       ACH, WT, SS);
    hipLaunchKernelGGL(s4_toeplitz, dim3(H_, 4), dim3(128), 0, stream,
                       KT, A2T, SS, T1, F(bb + 26), T2);
    hipLaunchKernelGGL(ln_apply, dim3(64), dim3(256), 0, stream, T2, T2, F(bb + 28), F(bb + 27));
    // gate, skip conv + BN, accumulate   (res branch is dead code -> skipped)
    hipLaunchKernelGGL(k_gate, dim3((unsigned)((nC + 255) / 256)), dim3(256), 0, stream, T2, GF);
    hipLaunchKernelGGL((conv_gemm<128, 128, 1>), g128, dim3(128), 0, stream,
                       GF, F(bb + 33), F(bb + 30), (const float*)nullptr, T1);
    hipLaunchKernelGGL(bn_stats, dim3(128), dim3(256), 0, stream, T1, 128, ST);
    hipLaunchKernelGGL(bn_apply, dim3((unsigned)((nC + 255) / 256)), dim3(256), 0, stream,
                       T1, T1, ST, F(bb + 32), F(bb + 31), 128, 0);
    hipLaunchKernelGGL(k_add, dim3((unsigned)((nC + 255) / 256)), dim3(256), 0, stream, SK, T1, nC);
  }

  // ---- head: skip/sqrt(NB) -> out1(+relu) -> out2 -> d_out ----
  hipLaunchKernelGGL(k_scale, dim3((unsigned)((nC + 255) / 256)), dim3(256), 0, stream,
                     SK, 0.40824829046386302f /* 1/sqrt(6) */, nC);
  hipLaunchKernelGGL((conv_gemm<128, 128, 1>), g128, dim3(128), 0, stream,
                     SK, F(212), F(209), (const float*)nullptr, T1);
  hipLaunchKernelGGL(bn_stats, dim3(128), dim3(256), 0, stream, T1, 128, ST);
  hipLaunchKernelGGL(bn_apply, dim3((unsigned)((nC + 255) / 256)), dim3(256), 0, stream,
                     T1, T1, ST, F(211), F(210), 128, 1);
  hipLaunchKernelGGL((conv_gemm<128, 32, 1>), g32, dim3(128), 0, stream,
                     T1, F(216), F(213), (const float*)nullptr, T2);
  hipLaunchKernelGGL(bn_stats, dim3(32), dim3(256), 0, stream, T2, 32, ST);
  const long nO = (long)B_ * 32 * L_;
  hipLaunchKernelGGL(bn_apply, dim3((unsigned)((nO + 255) / 256)), dim3(256), 0, stream,
                     T2, (float*)d_out, ST, F(215), F(214), 32, 0);
}
